// LuongAttention_6459630813727
// MI455X (gfx1250) — matrix operand order repaired and checked by reading, not compiled, not run on hardware
//
#include <hip/hip_runtime.h>
#include <hip/hip_bf16.h>
#include <math.h>

// Shapes
#define B_  8
#define TQ_ 64
#define TK_ 8192
#define H_  256

typedef __attribute__((ext_vector_type(16))) _Float16 v16h;
typedef __attribute__((ext_vector_type(8)))  _Float16 half8;
typedef __attribute__((ext_vector_type(8)))  float    v8f;

// Build a 16-half A/B fragment from two contiguous 8-half (16-byte) chunks.
__device__ __forceinline__ v16h make_frag16(const _Float16* lo_p, const _Float16* hi_p) {
    half8 lo = *(const half8*)lo_p;
    half8 hi = *(const half8*)hi_p;
    v16h a;
#pragma unroll
    for (int i = 0; i < 8; ++i) { a[i] = lo[i]; a[i + 8] = hi[i]; }
    return a;
}

// ---------------------------------------------------------------------------
// Kernel 1: Qt[b,q,h] = sum_o Q[b,q,o] * Wa[o,h]   (store f16)
//           qb[b,q]   = sum_o Q[b,q,o] * bias[o]   (store f32)
// One workgroup per (b,q) row; thread t computes column h = t.
// ---------------------------------------------------------------------------
__global__ void qt_prep_kernel(const float* __restrict__ query,
                               const float* __restrict__ Wa_w,
                               const float* __restrict__ Wa_b,
                               _Float16* __restrict__ Qt,
                               float* __restrict__ qb) {
    const int row = blockIdx.x;            // b*TQ + q
    const int t   = threadIdx.x;           // 0..255 = h
    __shared__ float qrow[H_];
    qrow[t] = query[(size_t)row * H_ + t];
    __syncthreads();

    float acc = 0.f;
#pragma unroll 8
    for (int o = 0; o < H_; ++o)
        acc = fmaf(qrow[o], Wa_w[o * H_ + t], acc);   // Wa_w[o][h], lanes coalesced over h
    Qt[(size_t)row * H_ + t] = (_Float16)acc;

    if (t == 0) {
        float s = 0.f;
#pragma unroll 8
        for (int o = 0; o < H_; ++o) s = fmaf(qrow[o], Wa_b[o], s);
        qb[row] = s;
    }
}

// ---------------------------------------------------------------------------
// Kernel 2: zero the context output region (split-K atomics accumulate into it)
// ---------------------------------------------------------------------------
__global__ void zero_ctx_kernel(float* __restrict__ ctx) {
    ctx[(size_t)blockIdx.x * 256 + threadIdx.x] = 0.f;
}

// ---------------------------------------------------------------------------
// Kernel 3: scores[b,q,k] = Qt[b,q,:]·keys[b,k,:] + qb[b,q]
// Grid: (TK/128, B). 256 threads = 8 waves; wave w handles 16 keys (N-tile).
// Qt (f16) staged in LDS; keys streamed from global, converted f32->f16.
// WMMA: M=64 (4 tiles) x N=16 x K=256 (8 steps of 32).
// ---------------------------------------------------------------------------
__global__ void scores_kernel(const _Float16* __restrict__ Qt,
                              const float* __restrict__ qb,
                              const float* __restrict__ keys,
                              float* __restrict__ outw) {
    const int b  = blockIdx.y;
    const int k0 = blockIdx.x * 128;
    const int t    = threadIdx.x;
    const int lane = t & 31, wave = t >> 5;
    const int lo   = lane & 15, hi = lane >> 4;

    __shared__ __align__(16) _Float16 sQt[TQ_ * H_];   // 32 KB
    __shared__ float sQb[TQ_];

    // Stage Qt[b] (64x256 f16) into LDS with 16-byte copies
    {
        const half8* src = (const half8*)(Qt + (size_t)b * TQ_ * H_);
        half8* dst = (half8*)sQt;
#pragma unroll
        for (int i = 0; i < (TQ_ * H_ / 8) / 256; ++i)
            dst[t + i * 256] = src[t + i * 256];
    }
    if (t < TQ_) sQb[t] = qb[b * TQ_ + t];
    __syncthreads();

    v8f acc[4];
#pragma unroll
    for (int m = 0; m < 4; ++m) acc[m] = (v8f){0.f,0.f,0.f,0.f,0.f,0.f,0.f,0.f};

    // keys slice for this wave's 16 key columns
    const float* kbase = keys + ((size_t)b * TK_ + k0 + wave * 16) * H_;

    for (int h0 = 0; h0 < H_; h0 += 32) {
        // B fragment: col N = lo (key), halves 0..7 -> K=h0+hi*8+e, 8..15 -> K+16
        v16h bfrag;
#pragma unroll
        for (int e = 0; e < 16; ++e)
            bfrag[e] = (_Float16)kbase[(size_t)lo * H_ + h0 + hi * 8 + (e < 8 ? e : e + 8)];

#pragma unroll
        for (int m = 0; m < 4; ++m) {
            // A fragment: row M = m*16+lo; halves 0..7 -> K=h0+hi*8+w, 8..15 -> K+16
            const _Float16* ap = sQt + (size_t)(m * 16 + lo) * H_ + h0 + hi * 8;
            v16h afrag = make_frag16(ap, ap + 16);
            acc[m] = __builtin_amdgcn_wmma_f32_16x16x32_f16(
                false, afrag, false, bfrag, (short)0, acc[m], false, false);
        }
    }

    // D layout: element v -> row M = 8*hi + v (+m*16), col N = lo
#pragma unroll
    for (int m = 0; m < 4; ++m) {
#pragma unroll
        for (int v = 0; v < 8; ++v) {
            const int qr = m * 16 + hi * 8 + v;
            const int kc = k0 + wave * 16 + lo;
            outw[((size_t)(b * TQ_ + qr)) * TK_ + kc] = acc[m][v] + sQb[qr];
        }
    }
}

// ---------------------------------------------------------------------------
// Kernel 4: in-place row softmax over TK=8192. One WG (256 thr) per (b,q) row.
// ---------------------------------------------------------------------------
__global__ void softmax_kernel(float* __restrict__ w) {
    const int t = threadIdx.x;
    float* p = w + (size_t)blockIdx.x * TK_;
    float x[32];
    float mx = -INFINITY;
#pragma unroll
    for (int j = 0; j < 32; ++j) { x[j] = p[t + j * 256]; mx = fmaxf(mx, x[j]); }

    __shared__ float red[256];
    red[t] = mx; __syncthreads();
#pragma unroll
    for (int s = 128; s > 0; s >>= 1) {
        if (t < s) red[t] = fmaxf(red[t], red[t + s]);
        __syncthreads();
    }
    mx = red[0]; __syncthreads();

    float sum = 0.f;
#pragma unroll
    for (int j = 0; j < 32; ++j) { x[j] = __expf(x[j] - mx); sum += x[j]; }
    red[t] = sum; __syncthreads();
#pragma unroll
    for (int s = 128; s > 0; s >>= 1) {
        if (t < s) red[t] += red[t + s];
        __syncthreads();
    }
    const float inv = 1.0f / red[0];
#pragma unroll
    for (int j = 0; j < 32; ++j) p[t + j * 256] = x[j] * inv;
}

// ---------------------------------------------------------------------------
// Kernel 5: context[b] += P[b](64x8192) @ keys[b](8192x256), split-K.
// Grid: (16, B); each WG handles K-chunk of 512; 8 waves x 2 N-tiles = N=256.
// f16 WMMA w/ f32 accumulation; atomicAdd f32 reduction across split-K.
// ---------------------------------------------------------------------------
__global__ void context_kernel(const float* __restrict__ weights,
                               const float* __restrict__ keys,
                               float* __restrict__ ctx) {
    const int b   = blockIdx.y;
    const int kc0 = blockIdx.x * 512;
    const int t    = threadIdx.x;
    const int lane = t & 31, wave = t >> 5;
    const int lo   = lane & 15, hi = lane >> 4;

    v8f acc[4][2];
#pragma unroll
    for (int m = 0; m < 4; ++m)
#pragma unroll
        for (int j = 0; j < 2; ++j)
            acc[m][j] = (v8f){0.f,0.f,0.f,0.f,0.f,0.f,0.f,0.f};

    for (int kk = kc0; kk < kc0 + 512; kk += 32) {
        // A fragments from softmax weights (f32 global -> f16)
        v16h afr[4];
#pragma unroll
        for (int m = 0; m < 4; ++m) {
            const float* wp = weights + (size_t)(b * TQ_ + m * 16 + lo) * TK_ + kk + hi * 8;
            float4 a0 = *(const float4*)(wp);
            float4 a1 = *(const float4*)(wp + 4);
            float4 a2 = *(const float4*)(wp + 16);
            float4 a3 = *(const float4*)(wp + 20);
            v16h a;
            a[0]=(_Float16)a0.x; a[1]=(_Float16)a0.y; a[2]=(_Float16)a0.z; a[3]=(_Float16)a0.w;
            a[4]=(_Float16)a1.x; a[5]=(_Float16)a1.y; a[6]=(_Float16)a1.z; a[7]=(_Float16)a1.w;
            a[8]=(_Float16)a2.x; a[9]=(_Float16)a2.y; a[10]=(_Float16)a2.z; a[11]=(_Float16)a2.w;
            a[12]=(_Float16)a3.x; a[13]=(_Float16)a3.y; a[14]=(_Float16)a3.z; a[15]=(_Float16)a3.w;
            afr[m] = a;
        }
        // B fragments from keys: col N = n0+lo, halves 0..7 -> K=kk+hi*8+s, 8..15 -> K+16
        v16h bfr[2];
#pragma unroll
        for (int j = 0; j < 2; ++j) {
            const int n0 = (wave * 2 + j) * 16;
            const float* kp = keys + ((size_t)b * TK_ + kk + hi * 8) * H_ + n0 + lo;
            v16h bb;
#pragma unroll
            for (int s = 0; s < 16; ++s)
                bb[s] = (_Float16)kp[(size_t)(s < 8 ? s : s + 8) * H_];
            bfr[j] = bb;
        }
#pragma unroll
        for (int m = 0; m < 4; ++m)
#pragma unroll
            for (int j = 0; j < 2; ++j)
                acc[m][j] = __builtin_amdgcn_wmma_f32_16x16x32_f16(
                    false, afr[m], false, bfr[j], (short)0, acc[m][j], false, false);
    }

#pragma unroll
    for (int m = 0; m < 4; ++m)
#pragma unroll
        for (int j = 0; j < 2; ++j)
#pragma unroll
            for (int v = 0; v < 8; ++v) {
                const int qr = m * 16 + hi * 8 + v;
                const int hc = (wave * 2 + j) * 16 + lo;
                atomicAdd(&ctx[(size_t)(b * TQ_ + qr) * H_ + hc], acc[m][j][v]);
            }
}

// ---------------------------------------------------------------------------
extern "C" void kernel_launch(void* const* d_in, const int* in_sizes, int n_in,
                              void* d_out, int out_size, void* d_ws, size_t ws_size,
                              hipStream_t stream) {
    const float* query = (const float*)d_in[0];   // [B,TQ,H]
    const float* keys  = (const float*)d_in[1];   // [B,TK,H]
    const float* Wa_w  = (const float*)d_in[2];   // [H,H]
    const float* Wa_b  = (const float*)d_in[3];   // [H]

    float* ctx     = (float*)d_out;                         // [B,TQ,H]
    float* weights = ctx + (size_t)B_ * TQ_ * H_;           // [B,TQ,TK]

    _Float16* Qt = (_Float16*)d_ws;                                   // B*TQ*H f16
    float*    qb = (float*)((char*)d_ws + (size_t)B_ * TQ_ * H_ * 2); // B*TQ f32

    qt_prep_kernel<<<B_ * TQ_, 256, 0, stream>>>(query, Wa_w, Wa_b, Qt, qb);
    zero_ctx_kernel<<<(B_ * TQ_ * H_) / 256, 256, 0, stream>>>(ctx);
    scores_kernel<<<dim3(TK_ / 128, B_), 256, 0, stream>>>(Qt, qb, keys, weights);
    softmax_kernel<<<B_ * TQ_, 256, 0, stream>>>(weights);
    context_kernel<<<dim3(16, B_), 256, 0, stream>>>(weights, keys, ctx);
}
